// FusedMSARowAttention_26620207300643
// MI455X (gfx1250) — compile-verified, matrix operand order checked by reading, not run on hardware
//
#include <hip/hip_runtime.h>

// ---------------- types ----------------
typedef __attribute__((ext_vector_type(16))) __bf16 v16bf;
typedef __attribute__((ext_vector_type(8)))  __bf16 v8bf;
typedef __attribute__((ext_vector_type(8)))  float  v8f;
typedef __attribute__((__vector_size__(16))) int i32x4;

#define NSEQ     128
#define SLEN     256
#define MSA_DIM  256
#define PAIR_DIM 128
#define N_HEADS  8
#define HEAD_DIM 32
#define MROWS    (NSEQ * SLEN)          // 32768
#define SCALE    0.17677669529663689f   // 1/sqrt(32)

// workspace layout (bytes)
#define OFF_WQKVT 0                       // bf16 [768][256]
#define OFF_WOT   (OFF_WQKVT + 768*256*2) // bf16 [256][256]
#define OFF_BIAS  (OFF_WOT   + 256*256*2) // f32  [8][256][256]
#define OFF_Q     (OFF_BIAS  + 8*256*256*4)       // bf16 [128*8][256][32]
#define OFF_K     (OFF_Q     + 1024*256*32*2)     // bf16 [128*8][256][32]
#define OFF_VT    (OFF_K     + 1024*256*32*2)     // bf16 [128*8][32][256]
#define OFF_O     (OFF_VT    + 1024*32*256*2)     // bf16 [32768][256]

#define PROW 260   // padded LDS row stride (floats) for the P slab
#define KROW 40    // padded LDS row stride (bf16) for staged K (32 + 8)
#define VROW 264   // padded LDS row stride (bf16) for staged V (256 + 8)

#if __has_builtin(__builtin_amdgcn_global_load_async_to_lds_b128) && \
    __has_builtin(__builtin_amdgcn_s_wait_asynccnt)
#define HAVE_ASYNC_LDS 1
// arg0: global source (AS1 int4*), arg1: LDS dest (AS3 int4*), imm offset, cpol
#define GL_ASYNC_B128(gsrc, ldst)                                   \
  __builtin_amdgcn_global_load_async_to_lds_b128(                   \
      (__attribute__((address_space(1))) i32x4*)(gsrc),             \
      (__attribute__((address_space(3))) i32x4*)(ldst), 0, 0)
#else
#define HAVE_ASYNC_LDS 0
#endif

// ---------------- helpers ----------------
__device__ __forceinline__ v8f wmma_bf16(v16bf a, v16bf b, v8f c) {
  return __builtin_amdgcn_wmma_f32_16x16x32_bf16(false, a, false, b, (short)0, c,
                                                 false, false);
}

// A/B fragment loader from f32 source: per-lane two contiguous runs of 8
// (K = k0..k0+7 -> elems 0..7, K = k0+16..k0+23 -> elems 8..15)
__device__ __forceinline__ v16bf frag_from_f32(const float* row, int k0) {
  const float4* p0 = reinterpret_cast<const float4*>(row + k0);
  const float4* p1 = reinterpret_cast<const float4*>(row + k0 + 16);
  float4 a0 = p0[0], a1 = p0[1], b0 = p1[0], b1 = p1[1];
  v16bf f;
  f[0]=(__bf16)a0.x; f[1]=(__bf16)a0.y; f[2]=(__bf16)a0.z; f[3]=(__bf16)a0.w;
  f[4]=(__bf16)a1.x; f[5]=(__bf16)a1.y; f[6]=(__bf16)a1.z; f[7]=(__bf16)a1.w;
  f[8]=(__bf16)b0.x; f[9]=(__bf16)b0.y; f[10]=(__bf16)b0.z; f[11]=(__bf16)b0.w;
  f[12]=(__bf16)b1.x; f[13]=(__bf16)b1.y; f[14]=(__bf16)b1.z; f[15]=(__bf16)b1.w;
  return f;
}

__device__ __forceinline__ v16bf frag_from_bf16(const __bf16* row, int k0) {
  v8bf u = *reinterpret_cast<const v8bf*>(row + k0);
  v8bf v = *reinterpret_cast<const v8bf*>(row + k0 + 16);
  v16bf f;
#pragma unroll
  for (int i = 0; i < 8; ++i) { f[i] = u[i]; f[8 + i] = v[i]; }
  return f;
}

// full-wave (32 lane) butterfly reductions via ds_swizzle (group-of-32 XOR)
#define SWZ_XOR(x, m) \
  __builtin_bit_cast(float, __builtin_amdgcn_ds_swizzle( \
      __builtin_bit_cast(int, (x)), 0x1f | ((m) << 10)))

__device__ __forceinline__ float wave_max(float v) {
  v = fmaxf(v, SWZ_XOR(v, 16));
  v = fmaxf(v, SWZ_XOR(v, 8));
  v = fmaxf(v, SWZ_XOR(v, 4));
  v = fmaxf(v, SWZ_XOR(v, 2));
  v = fmaxf(v, SWZ_XOR(v, 1));
  return v;
}
__device__ __forceinline__ float wave_sum(float v) {
  v += SWZ_XOR(v, 16);
  v += SWZ_XOR(v, 8);
  v += SWZ_XOR(v, 4);
  v += SWZ_XOR(v, 2);
  v += SWZ_XOR(v, 1);
  return v;
}

// ---------------- kernel 1: weight prep (f32 -> bf16, transposed) ----------
__global__ void prep_weights(const float* __restrict__ wqkv,
                             const float* __restrict__ wo,
                             __bf16* __restrict__ wqkvT,
                             __bf16* __restrict__ woT) {
  int idx = blockIdx.x * 256 + threadIdx.x;
  if (idx < 768 * 256) {              // wqkvT[n][k] = wqkv[k][n]
    int n = idx >> 8, k = idx & 255;
    wqkvT[idx] = (__bf16)wqkv[k * 768 + n];
  }
  int idx2 = idx - 768 * 256;
  if (idx2 >= 0 && idx2 < 256 * 256) { // woT[n][k] = wo[k][n]
    int n = idx2 >> 8, k = idx2 & 255;
    woT[idx2] = (__bf16)wo[k * 256 + n];
  }
}

// ---------------- kernel 2: pair bias  bias[h][i][j] = pair[i][j][:]·wpb[:,h]
__global__ void pair_bias_kernel(const float* __restrict__ pair,
                                 const float* __restrict__ wpb,
                                 float* __restrict__ bias) {
  __shared__ float wl[PAIR_DIM * N_HEADS];
  for (int i = threadIdx.x; i < PAIR_DIM * N_HEADS; i += 256) wl[i] = wpb[i];
  __syncthreads();
  int ij = blockIdx.x * 256 + threadIdx.x;  // (i,j) flat, 65536 total
  const float4* p4 =
      reinterpret_cast<const float4*>(pair + (size_t)ij * PAIR_DIM);
  float acc[N_HEADS] = {};
#pragma unroll 4
  for (int p = 0; p < PAIR_DIM / 4; ++p) {
    float4 v = p4[p];
#pragma unroll
    for (int h = 0; h < N_HEADS; ++h) {
      acc[h] += v.x * wl[(4 * p + 0) * N_HEADS + h];
      acc[h] += v.y * wl[(4 * p + 1) * N_HEADS + h];
      acc[h] += v.z * wl[(4 * p + 2) * N_HEADS + h];
      acc[h] += v.w * wl[(4 * p + 3) * N_HEADS + h];
    }
  }
#pragma unroll
  for (int h = 0; h < N_HEADS; ++h) bias[h * 65536 + ij] = acc[h];
}

// ---------------- kernel 3: QKV projection GEMM (WMMA bf16) ----------------
// C(32768x768) = msa(32768x256) x w_qkv(256x768); scatter into q/k/vT layouts
// 384 threads = 12 waves; each wave owns 4 N-tiles (12*4 = 48 = 768/16).
__global__ void __launch_bounds__(384)
qkv_gemm(const float* __restrict__ msa, const __bf16* __restrict__ wqkvT,
         __bf16* __restrict__ q, __bf16* __restrict__ k,
         __bf16* __restrict__ vT) {
  int mtile = blockIdx.x;              // 0..2047 (16 M-rows each)
  int wave = threadIdx.x >> 5;         // 0..11, each owns 4 N-tiles
  int lane = threadIdx.x & 31;
  int half = lane >> 4;
  int lm = lane & 15;
  const float* arow = msa + (size_t)(mtile * 16 + lm) * MSA_DIM;

  const __bf16* brow[4];
#pragma unroll
  for (int j = 0; j < 4; ++j)
    brow[j] = wqkvT + (size_t)((wave * 4 + j) * 16 + lm) * 256;

  v8f acc[4];
  v8f z = {};
#pragma unroll
  for (int j = 0; j < 4; ++j) acc[j] = z;

  for (int kb = 0; kb < 8; ++kb) {
    int k0 = kb * 32 + 8 * half;
    if (kb < 7) __builtin_prefetch(arow + k0 + 32, 0, 1);
    v16bf a = frag_from_f32(arow, k0);
#pragma unroll
    for (int j = 0; j < 4; ++j) {
      v16bf b = frag_from_bf16(brow[j], k0);
      acc[j] = wmma_bf16(a, b, acc[j]);
    }
  }

  int nrow = mtile >> 4;               // msa row index n
  int sbase = (mtile & 15) * 16;       // s base within row
#pragma unroll
  for (int j = 0; j < 4; ++j) {
    int e = (wave * 4 + j) * 16 + lm;  // output column 0..767
    int which = e >> 8;                // 0=q 1=k 2=v
    int head = (e >> 5) & 7;
    int d = e & 31;
    size_t rh = (size_t)(nrow * 8 + head);
#pragma unroll
    for (int r = 0; r < 8; ++r) {
      int s = sbase + r + 8 * half;
      __bf16 val = (__bf16)acc[j][r];
      if (which == 0)      q[(rh * 256 + s) * 32 + d] = val;
      else if (which == 1) k[(rh * 256 + s) * 32 + d] = val;
      else                 vT[(rh * 32 + d) * 256 + s] = val;
    }
  }
}

// ---------------- kernel 4: attention (scores + softmax + PV), WMMA --------
// grid: 128 rows * 8 heads * 2 q-halves = 2048 WGs; wave owns 16 q rows.
// K and V tiles for the (row, head) are staged once per WG into LDS
// (async-to-LDS when the toolchain exposes it), shared by all 8 waves.
__global__ void __launch_bounds__(256, 1)
attn_kernel(const __bf16* __restrict__ q, const __bf16* __restrict__ kk,
            const __bf16* __restrict__ vT, const float* __restrict__ bias,
            __bf16* __restrict__ O) {
  __shared__ __align__(16) float P[8 * 16 * PROW];
  __shared__ float rsum[8 * 16];
  __shared__ __align__(16) __bf16 Klds[256 * KROW];  // [s_k][d] padded
  __shared__ __align__(16) __bf16 Vlds[32 * VROW];   // [d][s_k] padded

  int wg = blockIdx.x;
  int qhalf = wg & 1;
  int rh = wg >> 1;                    // nrow*8 + head
  int head = rh & 7;
  int nrow = rh >> 3;
  int wave = threadIdx.x >> 5;
  int lane = threadIdx.x & 31;
  int half = lane >> 4;
  int lm = lane & 15;
  float* Pw = P + wave * 16 * PROW;
  int qbase = qhalf * 128 + wave * 16;

  // ---- stage K (256x32) and V^T (32x256) into LDS, 16B chunks ----
  {
    const __bf16* kg = kk + (size_t)rh * 256 * 32;
    const __bf16* vg = vT + (size_t)rh * 32 * 256;
    int t = threadIdx.x;               // 0..255
#pragma unroll
    for (int i = 0; i < 4; ++i) {
      int c = t + i * 256;             // chunk id 0..1023 (8 bf16 per chunk)
      int krow = c >> 2, kc8 = c & 3;  // K: 4 chunks per 32-wide row
      int vrow = c >> 5, vc8 = c & 31; // V: 32 chunks per 256-wide row
      const __bf16* kgp = kg + krow * 32 + kc8 * 8;
      const __bf16* vgp = vg + vrow * 256 + vc8 * 8;
      __bf16* klp = Klds + krow * KROW + kc8 * 8;
      __bf16* vlp = Vlds + vrow * VROW + vc8 * 8;
#if HAVE_ASYNC_LDS
      GL_ASYNC_B128(kgp, klp);
      GL_ASYNC_B128(vgp, vlp);
#else
      *(v8bf*)klp = *(const v8bf*)kgp;
      *(v8bf*)vlp = *(const v8bf*)vgp;
#endif
    }
#if HAVE_ASYNC_LDS
    __builtin_amdgcn_s_wait_asynccnt(0);
#endif
    __syncthreads();
  }

  // Q fragment (fixed across k-tiles): row = s_q, K-dim = head_dim (32)
  v16bf afrag =
      frag_from_bf16(q + ((size_t)rh * 256 + qbase + lm) * 32, 8 * half);

  // ---- scores: 16 k-tiles of 16, one bf16 WMMA each (K from LDS) ----
  const float* brow = bias + (size_t)head * 65536;
  for (int kt = 0; kt < 16; ++kt) {
    v16bf bfrag = frag_from_bf16(Klds + (kt * 16 + lm) * KROW, 8 * half);
    v8f c = {};
    c = wmma_bf16(afrag, bfrag, c);
    int sk = kt * 16 + lm;
#pragma unroll
    for (int r = 0; r < 8; ++r) {
      int mrow = r + 8 * half;
      int sq = qbase + mrow;
      Pw[mrow * PROW + sk] = c[r] * SCALE + brow[sq * 256 + sk];
    }
  }
  __syncthreads();

  // ---- softmax over k (rows of 256), wave-cooperative ----
  for (int r = 0; r < 16; ++r) {
    float* row = Pw + r * PROW;
    float v[8];
    float mx = -3.0e38f;
#pragma unroll
    for (int i = 0; i < 8; ++i) {
      v[i] = row[lane + 32 * i];
      mx = fmaxf(mx, v[i]);
    }
    mx = wave_max(mx);
    float s = 0.f;
#pragma unroll
    for (int i = 0; i < 8; ++i) {
      v[i] = __expf(v[i] - mx);
      s += v[i];
    }
    s = wave_sum(s);
#pragma unroll
    for (int i = 0; i < 8; ++i) row[lane + 32 * i] = v[i];
    if (lane == 0) rsum[wave * 16 + r] = s;
  }
  __syncthreads();

  // ---- PV: O(16x32) = P(16x256) x V(256x32), 8 K-chunks of 32 ----
  v8f o0 = {}, o1 = {};
  for (int kc = 0; kc < 8; ++kc) {
    int k0 = kc * 32 + 8 * half;
    v16bf pa = frag_from_f32(Pw + lm * PROW, k0);          // P as A-frag
    v16bf b0 = frag_from_bf16(Vlds + lm * VROW, k0);       // d = lm
    v16bf b1 = frag_from_bf16(Vlds + (lm + 16) * VROW, k0);// d = 16+lm
    o0 = wmma_bf16(pa, b0, o0);
    o1 = wmma_bf16(pa, b1, o1);
  }

  // ---- epilogue: divide by softmax denominator, store bf16 O ----
#pragma unroll
  for (int r = 0; r < 8; ++r) {
    int mrow = r + 8 * half;
    float inv = 1.0f / rsum[wave * 16 + mrow];
    int sq = qbase + mrow;
    size_t orow = ((size_t)nrow * 256 + sq) * 256 + head * 32;
    O[orow + lm] = (__bf16)(o0[r] * inv);
    O[orow + 16 + lm] = (__bf16)(o1[r] * inv);
  }
}

// ---------------- kernel 5: output projection GEMM (WMMA bf16) -------------
__global__ void out_gemm(const __bf16* __restrict__ O,
                         const __bf16* __restrict__ woT,
                         float* __restrict__ out) {
  int mtile = blockIdx.x;              // 0..2047
  int wave = threadIdx.x >> 5;         // each owns 2 N-tiles (16 total)
  int lane = threadIdx.x & 31;
  int half = lane >> 4;
  int lm = lane & 15;
  const __bf16* arow = O + (size_t)(mtile * 16 + lm) * 256;
  const __bf16* b0row = woT + (size_t)((wave * 2 + 0) * 16 + lm) * 256;
  const __bf16* b1row = woT + (size_t)((wave * 2 + 1) * 16 + lm) * 256;

  v8f acc0 = {}, acc1 = {};
  for (int kb = 0; kb < 8; ++kb) {
    int k0 = kb * 32 + 8 * half;
    v16bf a = frag_from_bf16(arow, k0);
    v16bf b0 = frag_from_bf16(b0row, k0);
    v16bf b1 = frag_from_bf16(b1row, k0);
    acc0 = wmma_bf16(a, b0, acc0);
    acc1 = wmma_bf16(a, b1, acc1);
  }
#pragma unroll
  for (int r = 0; r < 8; ++r) {
    size_t m = (size_t)mtile * 16 + r + 8 * half;
    out[m * 256 + (wave * 2 + 0) * 16 + lm] = acc0[r];
    out[m * 256 + (wave * 2 + 1) * 16 + lm] = acc1[r];
  }
}

// ---------------- launch ----------------
extern "C" void kernel_launch(void* const* d_in, const int* in_sizes, int n_in,
                              void* d_out, int out_size, void* d_ws,
                              size_t ws_size, hipStream_t stream) {
  (void)in_sizes; (void)n_in; (void)out_size; (void)ws_size;
  const float* msa  = (const float*)d_in[0];
  const float* pair = (const float*)d_in[1];
  const float* wqkv = (const float*)d_in[2];
  const float* wo   = (const float*)d_in[3];
  const float* wpb  = (const float*)d_in[4];
  float* out = (float*)d_out;

  char* ws = (char*)d_ws;
  __bf16* wqkvT = (__bf16*)(ws + OFF_WQKVT);
  __bf16* woT   = (__bf16*)(ws + OFF_WOT);
  float*  bias  = (float*)(ws + OFF_BIAS);
  __bf16* q     = (__bf16*)(ws + OFF_Q);
  __bf16* k     = (__bf16*)(ws + OFF_K);
  __bf16* vT    = (__bf16*)(ws + OFF_VT);
  __bf16* O     = (__bf16*)(ws + OFF_O);

  prep_weights<<<1024, 256, 0, stream>>>(wqkv, wo, wqkvT, woT);
  pair_bias_kernel<<<256, 256, 0, stream>>>(pair, wpb, bias);
  qkv_gemm<<<2048, 384, 0, stream>>>(msa, wqkvT, q, k, vT);
  attn_kernel<<<2048, 256, 0, stream>>>(q, k, vT, bias, O);
  out_gemm<<<2048, 256, 0, stream>>>(O, woT, out);
}
